// EV_GCN_65463891525762
// MI455X (gfx1250) — compile-verified, matrix operand order checked by reading, not run on hardware
//
#include <hip/hip_runtime.h>
#include <hip/hip_bf16.h>
#include <math.h>

// ---- problem constants (match reference) ----
#define N_NODES 2048
#define E_EDGES 65536
#define IN_DIM  2000
#define EDGE_DIM 120
#define HGC 128
#define LG1 4
#define WEIDU 64
#define NC 2
#define JK 512           // HGC*LG1
#define CLS_IN 1024      // 2*JK
#define BN_EPS 1e-5f
#define THR 0.5f

typedef __attribute__((ext_vector_type(16))) __bf16 v16bf;
typedef __attribute__((ext_vector_type(8)))  __bf16 v8bf;
typedef __attribute__((ext_vector_type(2)))  __bf16 v2bf;
typedef __attribute__((ext_vector_type(8)))  float  v8f;

#define BS_STRIDE 40   // bf16 elements per LDS row: 80B = 5*16B (aligned b128 reads,
                       // 20-bank row stride -> 16 lanes hit 16 distinct banks)

// ---------------------------------------------------------------------------
// GEMM: C[M,N] = A[M,K] @ B[K,N] (+bias, +relu), bf16 WMMA, f32 accum.
// Block = 128 thr (4 waves) -> 64x64 C tile; wave -> 16x64 (4 accs, shared A frag).
// B chunk (32K x 64N) staged via LDS: coalesced float4 global loads, stored
// transposed as packed bf16 pairs; fragments read as 2x ds_load_b128.
// Double-buffered, one barrier per K-chunk. Requires M%64==0, N%64==0.
// ---------------------------------------------------------------------------
__global__ void gemm_wmma_bf16(const float* __restrict__ A,
                               const float* __restrict__ B,
                               const float* __restrict__ bias,
                               float* __restrict__ C,
                               int M, int K, int N, int fuse_relu)
{
    __shared__ __bf16 Bs[2][64][BS_STRIDE];

    int tiles_m4 = M >> 6;                 // blocks along M
    int bm = blockIdx.x % tiles_m4;
    int bn = blockIdx.x / tiles_m4;
    int t    = threadIdx.x;
    int w    = t >> 5;
    int lane = t & 31;
    int hi   = lane >> 4;
    int r    = lane & 15;

    int tm = bm * 4 + w;
    const float* Arow = A + (size_t)(tm * 16 + r) * K;  // this lane's A row
    int bcol0 = bn * 64;

    // stage-stage indexing: thread -> (k-pair, 4-col group)
    int cq = (t & 15) * 4;     // column group base (0,4,...,60)
    int kp = t >> 4;           // k-pair id 0..7 (and +8 on 2nd iter)

    int nchunks = (K + 31) >> 5;

    v8f acc0 = {}, acc1 = {}, acc2 = {}, acc3 = {};

    // ---- B chunk -> LDS (transposed, bf16) ----
    auto stage = [&](int chunk, int buf) {
        int kbase = chunk * 32;
        bool full = (kbase + 32 <= K);
#pragma unroll
        for (int i = 0; i < 2; ++i) {
            int klocal = 2 * (kp + 8 * i);
            int kk = kbase + klocal;
            float4 fa = make_float4(0.f, 0.f, 0.f, 0.f);
            float4 fb = make_float4(0.f, 0.f, 0.f, 0.f);
            if (full) {
                fa = *(const float4*)(B + (size_t)kk * N + bcol0 + cq);
                fb = *(const float4*)(B + (size_t)(kk + 1) * N + bcol0 + cq);
            } else {
                if (kk < K)     fa = *(const float4*)(B + (size_t)kk * N + bcol0 + cq);
                if (kk + 1 < K) fb = *(const float4*)(B + (size_t)(kk + 1) * N + bcol0 + cq);
            }
            v2bf p0; p0[0] = (__bf16)fa.x; p0[1] = (__bf16)fb.x;
            v2bf p1; p1[0] = (__bf16)fa.y; p1[1] = (__bf16)fb.y;
            v2bf p2; p2[0] = (__bf16)fa.z; p2[1] = (__bf16)fb.z;
            v2bf p3; p3[0] = (__bf16)fa.w; p3[1] = (__bf16)fb.w;
            *(v2bf*)&Bs[buf][cq + 0][klocal] = p0;
            *(v2bf*)&Bs[buf][cq + 1][klocal] = p1;
            *(v2bf*)&Bs[buf][cq + 2][klocal] = p2;
            *(v2bf*)&Bs[buf][cq + 3][klocal] = p3;
        }
    };

    // ---- B fragment from LDS: two aligned 16B runs ----
    auto frag = [&](int buf, int colLocal) -> v16bf {
        v8bf lo = *(const v8bf*)&Bs[buf][colLocal][hi * 8];
        v8bf hi8 = *(const v8bf*)&Bs[buf][colLocal][16 + hi * 8];
        return __builtin_shufflevector(lo, hi8, 0, 1, 2, 3, 4, 5, 6, 7,
                                       8, 9, 10, 11, 12, 13, 14, 15);
    };

    stage(0, 0);
    __syncthreads();

    for (int ch = 0; ch < nchunks; ++ch) {
        int buf = ch & 1;
        if (ch + 1 < nchunks) stage(ch + 1, buf ^ 1);

        // A fragment (direct global, float2 pairs)
        int kbase = ch * 32;
        bool full = (kbase + 32 <= K);
        v16bf a;
        if (full) {
#pragma unroll
            for (int v = 0; v < 8; ++v) {
                int kk = kbase + ((v >> 2) << 4) + (hi << 3) + ((v & 3) << 1);
                float2 av = *(const float2*)(Arow + kk);
                a[2 * v]     = (__bf16)av.x;
                a[2 * v + 1] = (__bf16)av.y;
            }
        } else {
#pragma unroll
            for (int v = 0; v < 8; ++v) {
                int kk = kbase + ((v >> 2) << 4) + (hi << 3) + ((v & 3) << 1);
                float a0 = (kk     < K) ? Arow[kk]     : 0.f;
                float a1 = (kk + 1 < K) ? Arow[kk + 1] : 0.f;
                a[2 * v]     = (__bf16)a0;
                a[2 * v + 1] = (__bf16)a1;
            }
        }

        v16bf b0 = frag(buf, r);
        v16bf b1 = frag(buf, r + 16);
        v16bf b2 = frag(buf, r + 32);
        v16bf b3 = frag(buf, r + 48);
        acc0 = __builtin_amdgcn_wmma_f32_16x16x32_bf16(false, a, false, b0, (short)0, acc0, false, false);
        acc1 = __builtin_amdgcn_wmma_f32_16x16x32_bf16(false, a, false, b1, (short)0, acc1, false, false);
        acc2 = __builtin_amdgcn_wmma_f32_16x16x32_bf16(false, a, false, b2, (short)0, acc2, false, false);
        acc3 = __builtin_amdgcn_wmma_f32_16x16x32_bf16(false, a, false, b3, (short)0, acc3, false, false);

        __syncthreads();
    }

    int bcol = bcol0 + r;
#pragma unroll
    for (int v = 0; v < 8; ++v) {
        int orow = tm * 16 + (hi << 3) + v;   // C/D: VGPR v -> row v (+8 for hi half)
        float* Crow = C + (size_t)orow * N + bcol;
        float x0 = acc0[v], x1 = acc1[v], x2 = acc2[v], x3 = acc3[v];
        if (bias) {
            const float* bp = bias + bcol;
            x0 += bp[0]; x1 += bp[16]; x2 += bp[32]; x3 += bp[48];
        }
        if (fuse_relu) {
            x0 = fmaxf(x0, 0.f); x1 = fmaxf(x1, 0.f);
            x2 = fmaxf(x2, 0.f); x3 = fmaxf(x3, 0.f);
        }
        Crow[0] = x0; Crow[16] = x1; Crow[32] = x2; Crow[48] = x3;
    }
}

// ---------------------------------------------------------------------------
// PAE tail: w = sigmoid(Hpae[e,:] @ w2 + b2); emit merged, wt, keep.
// ---------------------------------------------------------------------------
__global__ void pae_finish_kernel(const float* __restrict__ Hpae,
                                  const float* __restrict__ w2, const float* __restrict__ b2,
                                  float* __restrict__ merged, float* __restrict__ wt,
                                  float* __restrict__ keep)
{
    int e = blockIdx.x * blockDim.x + threadIdx.x;
    if (e >= E_EDGES) return;
    const float* row = Hpae + (size_t)e * WEIDU;
    float s = b2[0];
#pragma unroll 8
    for (int j = 0; j < WEIDU; ++j) s = fmaf(row[j], w2[j], s);
    float w = 1.f / (1.f + __expf(-s));
    merged[2 * e]     = 1.f - w;
    merged[2 * e + 1] = w;
    float k = (w > THR) ? 1.f : 0.f;
    wt[e]   = w * k;
    keep[e] = k;
}

// val[n] = sum_e mask[n,e] * wt[e] : one block per node row (streams 64KB/row)
__global__ void val_kernel(const unsigned char* __restrict__ mask,
                           const float* __restrict__ wt, float* __restrict__ val)
{
    __shared__ float red[256];
    int n = blockIdx.x;
    float s = 0.f;
    for (int e = threadIdx.x; e < E_EDGES; e += blockDim.x)
        if (mask[(size_t)n * E_EDGES + e]) s += wt[e];
    red[threadIdx.x] = s;
    __syncthreads();
    for (int o = 128; o > 0; o >>= 1) {
        if ((int)threadIdx.x < o) red[threadIdx.x] += red[threadIdx.x + o];
        __syncthreads();
    }
    if (threadIdx.x == 0) val[n] = red[0];
}

__global__ void zero_kernel(float* __restrict__ p, int n)
{
    int i = blockIdx.x * blockDim.x + threadIdx.x;
    if (i < n) p[i] = 0.f;
}

__global__ void ei_kernel(const int* __restrict__ ei, float* __restrict__ out, int n)
{
    int i = blockIdx.x * blockDim.x + threadIdx.x;
    if (i < n) out[i] = (float)ei[i];
}

// den[dst] += keep
__global__ void den_kernel(const int* __restrict__ ei, const float* __restrict__ keep,
                           float* __restrict__ den)
{
    int e = blockIdx.x * blockDim.x + threadIdx.x;
    if (e >= E_EDGES) return;
    float k = keep[e];
    if (k != 0.f) atomicAdd(&den[ei[E_EDGES + e]], k);
}

// num[dst,:] += Yl[src,:] for kept edges; one block (128 thr) per edge
__global__ void scatter_kernel(const int* __restrict__ ei, const float* __restrict__ keep,
                               const float* __restrict__ Yl, float* __restrict__ num)
{
    int e = blockIdx.x;
    int f = threadIdx.x;
    if (keep[e] != 0.f) {
        int src = ei[e];
        int dst = ei[E_EDGES + e];
        atomicAdd(&num[(size_t)dst * HGC + f], Yl[(size_t)src * HGC + f]);
    }
}

// H = relu(num/den + Yr + b); also writes JK slice
__global__ void combine_kernel(const float* __restrict__ num, const float* __restrict__ den,
                               const float* __restrict__ Yr, const float* __restrict__ b,
                               float* __restrict__ H, float* __restrict__ slice)
{
    int i = blockIdx.x * blockDim.x + threadIdx.x;
    if (i >= N_NODES * HGC) return;
    int n = i / HGC, f = i % HGC;
    float d   = den[n];
    float agg = (d > 0.f) ? num[i] / fmaxf(d, 1.f) : 0.f;
    float h   = fmaxf(agg + Yr[i] + b[f], 0.f);
    H[i] = h;
    slice[(size_t)n * JK + f] = h;
}

__global__ void diff_kernel(const float* __restrict__ L, const float* __restrict__ R,
                            float* __restrict__ D, int n)
{
    int i = blockIdx.x * blockDim.x + threadIdx.x;
    if (i < n) D[i] = L[i] - R[i];
}

// BN(eval) + final 256->2 linear; optional lab=argmax
__global__ void head_kernel(const float* __restrict__ X,
                            const float* __restrict__ gamma, const float* __restrict__ beta,
                            const float* __restrict__ w2, const float* __restrict__ b2,
                            float* __restrict__ out, float* __restrict__ lab)
{
    int n = blockIdx.x * blockDim.x + threadIdx.x;
    if (n >= N_NODES) return;
    float inv = rsqrtf(1.f + BN_EPS);
    float c0 = b2[0], c1 = b2[1];
    for (int j = 0; j < 256; ++j) {
        float g = X[(size_t)n * 256 + j] * gamma[j] * inv + beta[j];
        c0 = fmaf(g, w2[j * NC + 0], c0);
        c1 = fmaf(g, w2[j * NC + 1], c1);
    }
    out[n * NC]     = c0;
    out[n * NC + 1] = c1;
    if (lab) lab[n] = (c1 > c0) ? 1.f : 0.f;   // argmax (ties -> 0)
}

// CONC = [ |L-R| , L*lab + R*(1-lab) ]
__global__ void conc_kernel(const float* __restrict__ L, const float* __restrict__ R,
                            const float* __restrict__ lab, float* __restrict__ C)
{
    int i = blockIdx.x * blockDim.x + threadIdx.x;
    if (i >= N_NODES * CLS_IN) return;
    int n = i / CLS_IN, k = i % CLS_IN;
    if (k < JK) {
        C[i] = fabsf(L[(size_t)n * JK + k] - R[(size_t)n * JK + k]);
    } else {
        int j = k - JK;
        float l = lab[n];
        C[i] = L[(size_t)n * JK + j] * l + R[(size_t)n * JK + j] * (1.f - l);
    }
}

// ---------------------------------------------------------------------------
extern "C" void kernel_launch(void* const* d_in, const int* in_sizes, int n_in,
                              void* d_out, int out_size, void* d_ws, size_t ws_size,
                              hipStream_t stream)
{
    const float* f_dis = (const float*)d_in[0];
    const float* f_hea = (const float*)d_in[1];
    const float* enet  = (const float*)d_in[2];
    const int*   eidx  = (const int*)d_in[3];
    const unsigned char* mask = (const unsigned char*)d_in[4];
    const float* pae_w1 = (const float*)d_in[5];
    const float* pae_b1 = (const float*)d_in[6];
    const float* pae_w2 = (const float*)d_in[7];
    const float* pae_b2 = (const float*)d_in[8];
    const float* sage0_wl = (const float*)d_in[9];
    const float* sage0_wr = (const float*)d_in[10];
    const float* sage0_b  = (const float*)d_in[11];
    const float* sage_wl  = (const float*)d_in[12];   // [3,128,128]
    const float* sage_wr  = (const float*)d_in[13];
    const float* sage_b   = (const float*)d_in[14];   // [3,128]
    const float* cls_w1 = (const float*)d_in[15];
    const float* cls_b1 = (const float*)d_in[16];
    const float* cls_g  = (const float*)d_in[17];
    const float* cls_be = (const float*)d_in[18];
    const float* cls_w2 = (const float*)d_in[19];
    const float* cls_b2 = (const float*)d_in[20];
    const float* ecls_w1 = (const float*)d_in[21];
    const float* ecls_b1 = (const float*)d_in[22];
    const float* ecls_g  = (const float*)d_in[23];
    const float* ecls_be = (const float*)d_in[24];
    const float* ecls_w2 = (const float*)d_in[25];
    const float* ecls_b2 = (const float*)d_in[26];

    // output layout (flat, return order)
    float* out        = (float*)d_out;
    float* out_logit  = out;                         // [2048,2]
    float* out_merged = out + 4096;                  // [65536,2]
    float* out_elogit = out + 4096 + 131072;         // [2048,2]
    float* out_val    = out + 4096 + 131072 + 4096;  // [2048]
    float* out_ei     = out_val + 2048;              // [2,65536] as float
    float* out_wt     = out_ei + 2 * E_EDGES;        // [65536]

    // workspace layout
    float* ws = (float*)d_ws;
    size_t o = 0;
    float* keep = ws + o; o += E_EDGES;
    float* den  = ws + o; o += N_NODES;
    float* Yl   = ws + o; o += (size_t)N_NODES * HGC;
    float* Yr   = ws + o; o += (size_t)N_NODES * HGC;
    float* num  = ws + o; o += (size_t)N_NODES * HGC;
    float* Ha   = ws + o; o += (size_t)N_NODES * HGC;
    float* Hb   = ws + o; o += (size_t)N_NODES * HGC;
    float* Lb   = ws + o; o += (size_t)N_NODES * JK;
    float* Rb   = ws + o; o += (size_t)N_NODES * JK;
    float* Db   = ws + o; o += (size_t)N_NODES * JK;
    float* E1   = ws + o; o += (size_t)N_NODES * 256;
    float* CONC = ws + o; o += (size_t)N_NODES * CLS_IN;
    float* C1   = ws + o; o += (size_t)N_NODES * 256;
    float* lab  = ws + o; o += N_NODES;
    float* Hpae = ws + o; o += (size_t)E_EDGES * WEIDU;   // [65536,64]

    auto gemm = [&](const float* A, const float* B, const float* bias, float* C,
                    int M, int K, int Nn, int relu) {
        int blocks = (M / 64) * (Nn / 64);   // one 64x64 tile per 128-thread block
        gemm_wmma_bf16<<<blocks, 128, 0, stream>>>(A, B, bias, C, M, K, Nn, relu);
    };

    // 1) PAE edge net: hidden layer on WMMA, then sigmoid tail
    gemm(enet, pae_w1, pae_b1, Hpae, E_EDGES, EDGE_DIM, WEIDU, 1);
    pae_finish_kernel<<<E_EDGES / 256, 256, 0, stream>>>(Hpae, pae_w2, pae_b2,
                                                         out_merged, out_wt, keep);
    // 2) val = mask @ wt
    val_kernel<<<N_NODES, 256, 0, stream>>>(mask, out_wt, out_val);
    // 3) edge_index passthrough (as float)
    ei_kernel<<<(2 * E_EDGES) / 256, 256, 0, stream>>>(eidx, out_ei, 2 * E_EDGES);
    // 4) degree of kept in-edges (shared by all layers/branches)
    zero_kernel<<<(N_NODES + 255) / 256, 256, 0, stream>>>(den, N_NODES);
    den_kernel<<<E_EDGES / 256, 256, 0, stream>>>(eidx, keep, den);

    // 5) SAGE branches (linear commuted with mean-aggregation: agg(X)@W == agg(X@W))
    for (int br = 0; br < 2; ++br) {
        const float* X = (br == 0) ? f_dis : f_hea;
        float* Jbuf = (br == 0) ? Lb : Rb;
        int Kdim = IN_DIM;
        for (int layer = 0; layer < LG1; ++layer) {
            const float* wl = (layer == 0) ? sage0_wl : sage_wl + (size_t)(layer - 1) * HGC * HGC;
            const float* wr = (layer == 0) ? sage0_wr : sage_wr + (size_t)(layer - 1) * HGC * HGC;
            const float* bb = (layer == 0) ? sage0_b  : sage_b  + (size_t)(layer - 1) * HGC;
            gemm(X, wl, nullptr, Yl, N_NODES, Kdim, HGC, 0);
            gemm(X, wr, nullptr, Yr, N_NODES, Kdim, HGC, 0);
            zero_kernel<<<(N_NODES * HGC + 255) / 256, 256, 0, stream>>>(num, N_NODES * HGC);
            scatter_kernel<<<E_EDGES, HGC, 0, stream>>>(eidx, keep, Yl, num);
            float* Hcur = (layer & 1) ? Hb : Ha;
            combine_kernel<<<(N_NODES * HGC + 255) / 256, 256, 0, stream>>>(
                num, den, Yr, bb, Hcur, Jbuf + (size_t)layer * HGC);
            X = Hcur;
            Kdim = HGC;
        }
    }

    // 6) edge-position head on L-R
    diff_kernel<<<(N_NODES * JK + 255) / 256, 256, 0, stream>>>(Lb, Rb, Db, N_NODES * JK);
    gemm(Db, ecls_w1, ecls_b1, E1, N_NODES, JK, 256, 1);
    head_kernel<<<(N_NODES + 255) / 256, 256, 0, stream>>>(E1, ecls_g, ecls_be,
                                                           ecls_w2, ecls_b2, out_elogit, lab);
    // 7) main classifier on [ |L-R| , dis ]
    conc_kernel<<<(N_NODES * CLS_IN + 255) / 256, 256, 0, stream>>>(Lb, Rb, lab, CONC);
    gemm(CONC, cls_w1, cls_b1, C1, N_NODES, CLS_IN, 256, 1);
    head_kernel<<<(N_NODES + 255) / 256, 256, 0, stream>>>(C1, cls_g, cls_be,
                                                           cls_w2, cls_b2, out_logit, nullptr);
}
